// SA_Layer_45603962749650
// MI455X (gfx1250) — matrix operand order repaired and checked by reading, not compiled, not run on hardware
//
#include <hip/hip_runtime.h>
#include <hip/hip_bf16.h>

// ---------------------------------------------------------------------------
// Problem constants (match reference): B=8, S=2048, E=512, D=512
// ---------------------------------------------------------------------------
constexpr int kB = 8;
constexpr int kS = 2048;
constexpr int kE = 512;
constexpr int kD = 512;
constexpr float kScale = 0.044194173824159216f;  // 1/sqrt(512)

typedef __attribute__((ext_vector_type(16))) __bf16        v16bf;
typedef __attribute__((ext_vector_type(8)))  __bf16        v8bf;
typedef __attribute__((ext_vector_type(8)))  float         v8f;
typedef __attribute__((ext_vector_type(4)))  float         v4f;
typedef __attribute__((ext_vector_type(4)))  unsigned int  v4u;
typedef __attribute__((ext_vector_type(2)))  unsigned long long v2u64;
typedef __attribute__((ext_vector_type(4)))  unsigned int  v4u32;
typedef __attribute__((ext_vector_type(8)))  int           v8i32;
typedef __attribute__((ext_vector_type(4)))  int           v4i32;

// LDS row pitch: 40 bf16 = 80 bytes -> 16B-aligned rows, 64-bank friendly b128
constexpr int kP = 40;

// ---------------------------------------------------------------------------
// Tensor Data Mover staging (gfx1250).  Descriptor built ONCE per kernel;
// per-issue we patch only lds_addr + global_addr (g0[1..3]).
//  - data_size = 2B (bf16), tile = rows x 32 elems, tensor = 2048 x 512,
//    dim0 stride = 512 elems
//  - pad_enable, pad_interval code 3 (16 DWORDs), pad_amount code 3 (4 DWORDs)
//    -> reproduces the 40-elem LDS pitch in hardware
// ---------------------------------------------------------------------------
#if __has_builtin(__builtin_amdgcn_tensor_load_to_lds) && \
    __has_builtin(__builtin_amdgcn_s_wait_tensorcnt)
#define HAVE_TDM 1
#if __has_include(<hip/amd_detail/amd_gfx1250_TDM.h>)
#define TDM_ARGS6 1
#endif
#endif

#if defined(HAVE_TDM)
__device__ __forceinline__ unsigned lds_off_of(const void* p) {
  return (unsigned)(size_t)(__attribute__((address_space(3))) const void*)p;
}

struct TdmDesc {
  v4u32 g0;
  v8i32 g1;
};

__device__ __forceinline__ TdmDesc tdm_make_desc(int tileRows) {
  TdmDesc d;
  d.g0[0] = 1u;          // count=1 (valid user D#)
  d.g0[1] = 0u;          // lds_addr (patched per issue)
  d.g0[2] = 0u;          // global_addr lo (patched)
  d.g0[3] = 2u << 30;    // type=2 (patched with addr hi)
  d.g1[0] = (1 << 16)    // data_size = 2 bytes
          | (1 << 20)    // pad_enable
          | (3 << 22)    // pad_interval: 16 DWORDs (one 64B tile row)
          | (3 << 25);   // pad_amount: 4 DWORDs (16B)
  d.g1[1] = kD << 16;    // tensor_dim0 = 512
  d.g1[2] = kS << 16;    // tensor_dim1 = 2048
  d.g1[3] = 32 << 16;    // tile_dim0 = 32 elems
  d.g1[4] = tileRows;    // tile_dim1
  d.g1[5] = kD;          // tensor_dim0_stride = 512
  d.g1[6] = 0;
  d.g1[7] = 0;
  return d;
}

__device__ __forceinline__ void tdm_issue(TdmDesc d, unsigned ldsOff,
                                          const __bf16* gSrc) {
  const unsigned long long ga = (unsigned long long)(size_t)gSrc;
  d.g0[1] = ldsOff;
  d.g0[2] = (unsigned)(ga & 0xffffffffu);
  d.g0[3] = (unsigned)((ga >> 32) & 0x1ffffffu) | (2u << 30);
  const v4i32 z4 = {};
#if defined(TDM_ARGS6)
  const v8i32 z8 = {};
  __builtin_amdgcn_tensor_load_to_lds(d.g0, d.g1, z4, z4, z8, 0);
#else
  __builtin_amdgcn_tensor_load_to_lds(d.g0, d.g1, z4, z4, 0);
#endif
}
#endif  // HAVE_TDM

// ---------------------------------------------------------------------------
// WMMA 16x16x32 bf16 fragment loaders (wave32 layouts, ISA 7.12.2)
// ---------------------------------------------------------------------------
__device__ __forceinline__ v16bf load_a_frag(const __bf16* row, int half) {
  v8bf lo = *(const v8bf*)(row + half * 8);
  v8bf hi = *(const v8bf*)(row + 16 + half * 8);
  return __builtin_shufflevector(lo, hi, 0, 1, 2, 3, 4, 5, 6, 7,
                                 8, 9, 10, 11, 12, 13, 14, 15);
}
__device__ __forceinline__ v16bf load_bT_frag(const __bf16* row, int half) {
  v8bf lo = *(const v8bf*)(row + half * 8);
  v8bf hi = *(const v8bf*)(row + 16 + half * 8);
  return __builtin_shufflevector(lo, hi, 0, 1, 2, 3, 4, 5, 6, 7,
                                 8, 9, 10, 11, 12, 13, 14, 15);
}

__device__ __forceinline__ v8f wmma_bf16(const v16bf& a, const v16bf& b, const v8f& c) {
  return __builtin_amdgcn_wmma_f32_16x16x32_bf16(false, a, false, b, (short)0, c,
                                                 false, false);
}

__device__ __forceinline__ unsigned long long pack4_bf16(v4f f) {
  union { __bf16 h[4]; unsigned long long u; } u;
#pragma unroll
  for (int i = 0; i < 4; ++i) u.h[i] = (__bf16)f[i];
  return u.u;
}

// ---------------------------------------------------------------------------
// Kernel 0: f32 -> bf16, 8 elements / thread, 16B stores
// ---------------------------------------------------------------------------
__global__ __launch_bounds__(256) void f32_to_bf16_kernel(const float* __restrict__ in,
                                                          __bf16* __restrict__ out) {
  const int i = blockIdx.x * 256 + threadIdx.x;
  const v4f* p = (const v4f*)in;
  v2u64 pkt;
  pkt[0] = pack4_bf16(p[2 * i]);
  pkt[1] = pack4_bf16(p[2 * i + 1]);
  *(v2u64*)(out + (size_t)i * 8) = pkt;
}

// ---------------------------------------------------------------------------
// Kernel 1: out = bf16(relu(x @ W + bias))  (f32 sources -> manual staging)
// grid = (S/128, D/64, B), block = 256 (8 wave32)
// ---------------------------------------------------------------------------
__global__ __launch_bounds__(256) void proj_bias_relu_kernel(
    const float* __restrict__ x, const float* __restrict__ W,
    const float* __restrict__ bias, __bf16* __restrict__ out) {
  __shared__ __bf16 As[128][kP];
  __shared__ __bf16 Bw[64][kP];

  const int b = blockIdx.z;
  const int m0 = blockIdx.x * 128;
  const int n0 = blockIdx.y * 64;
  const int tid = threadIdx.x;
  const int wave = tid >> 5, lane = tid & 31;
  const int half = lane >> 4, lcol = lane & 15;
  const float* xb = x + (size_t)b * kS * kE;

  v8f acc[4] = {};
  for (int kt = 0; kt < kE; kt += 32) {
#pragma unroll
    for (int it = 0; it < 4; ++it) {
      const int j = tid + it * 256;
      const int r = j >> 3, c = (j & 7) * 4;
      v4f f = *(const v4f*)&xb[(size_t)(m0 + r) * kE + kt + c];
      *(unsigned long long*)&As[r][c] = pack4_bf16(f);
    }
#pragma unroll
    for (int it = 0; it < 2; ++it) {
      const int j = tid + it * 256;
      const int r = j >> 4, c = (j & 15) * 4;
      v4f w = *(const v4f*)&W[(size_t)(kt + r) * kD + n0 + c];
#pragma unroll
      for (int i = 0; i < 4; ++i) Bw[c + i][r] = (__bf16)w[i];
    }
    if (kt + 32 < kE)
      __builtin_prefetch(&xb[(size_t)(m0 + (tid >> 1)) * kE + kt + 32], 0, 1);
    __syncthreads();

    const v16bf a = load_a_frag(&As[wave * 16 + lcol][0], half);
#pragma unroll
    for (int t = 0; t < 4; ++t) {
      const v16bf bf = load_bT_frag(&Bw[t * 16 + lcol][0], half);
      acc[t] = wmma_bf16(a, bf, acc[t]);
    }
    __syncthreads();
  }

#pragma unroll
  for (int t = 0; t < 4; ++t) {
    const int col = n0 + t * 16 + lcol;
    const float bv = bias[col];
#pragma unroll
    for (int r = 0; r < 8; ++r) {
      const int row = m0 + wave * 16 + r + half * 8;
      float v = acc[t][r] + bv;
      v = v > 0.f ? v : 0.f;
      out[((size_t)b * kS + row) * kD + col] = (__bf16)v;
    }
  }
}

// ---------------------------------------------------------------------------
// Kernel 2: per-key (column) softmax stats over the query axis.
// grid = (S/64 key tiles, B), block = 256.
// TDM double-buffered staging of Q/K tiles; WMMA for score tiles.
// ---------------------------------------------------------------------------
__global__ __launch_bounds__(256) void col_softmax_stats_kernel(
    const __bf16* __restrict__ qb, const __bf16* __restrict__ kb,
    float* __restrict__ colmax, float* __restrict__ colrcp) {
  __shared__ __bf16 Qs[2][128][kP];
  __shared__ __bf16 Ks[2][64][kP];
  __shared__ float redm[8][64];
  __shared__ float redl[8][64];

  const int b = blockIdx.y;
  const int n0 = blockIdx.x * 64;  // key base
  const int tid = threadIdx.x;
  const int wave = tid >> 5, lane = tid & 31;
  const int half = lane >> 4, lcol = lane & 15;
  const __bf16* qB = qb + (size_t)b * kS * kD;
  const __bf16* kB = kb + (size_t)b * kS * kD;

  float m_run[4], l_run[4];
#pragma unroll
  for (int t = 0; t < 4; ++t) { m_run[t] = -1.0e30f; l_run[t] = 0.f; }

  constexpr int kIters = (kS / 128) * (kD / 32);  // 256, flattened (q0, dt)
  v8f acc[4];

#if defined(HAVE_TDM)
  const TdmDesc dQ = tdm_make_desc(128);
  const TdmDesc dK = tdm_make_desc(64);
  const unsigned qsOff[2] = {lds_off_of(&Qs[0][0][0]), lds_off_of(&Qs[1][0][0])};
  const unsigned ksOff[2] = {lds_off_of(&Ks[0][0][0]), lds_off_of(&Ks[1][0][0])};
  const __bf16* kBase = kB + (size_t)n0 * kD;
  const __bf16* qT = qB;    // current tile pointer (advances incrementally)
  const __bf16* kT = kBase;
  if (tid < 32) {  // prologue: tile 0 -> buffer 0
    tdm_issue(dQ, qsOff[0], qT);
    tdm_issue(dK, ksOff[0], kT);
  }
#endif

  for (int it = 0; it < kIters; ++it) {
    const int cur = it & 1;
#if defined(HAVE_TDM)
    const bool wrap = (it & 15) == 15;  // end of one 128-query strip
    const __bf16* qTn = qT + (wrap ? (size_t)(128 * kD - 480) : 32);
    const __bf16* kTn = wrap ? kBase : kT + 32;
    if (tid < 32) {
      if (it + 1 < kIters) {  // prefetch next tile into alternate buffer
        tdm_issue(dQ, qsOff[cur ^ 1], qTn);
        tdm_issue(dK, ksOff[cur ^ 1], kTn);
        __builtin_amdgcn_s_wait_tensorcnt(2);  // current buffer's 2 loads done
      } else {
        __builtin_amdgcn_s_wait_tensorcnt(0);
      }
    }
    qT = qTn;
    kT = kTn;
#else
    const int q0 = (it >> 4) * 128;
    const int dt = (it & 15) * 32;
#pragma unroll
    for (int c2 = 0; c2 < 2; ++c2) {
      const int j = tid + c2 * 256;
      const int r = j >> 2, c = (j & 3) * 8;
      *(v4u*)&Qs[cur][r][c] = *(const v4u*)&qB[(size_t)(q0 + r) * kD + dt + c];
    }
    {
      const int r = tid >> 2, c = (tid & 3) * 8;
      *(v4u*)&Ks[cur][r][c] = *(const v4u*)&kB[(size_t)(n0 + r) * kD + dt + c];
    }
#endif
    __syncthreads();

    if ((it & 15) == 0) {
#pragma unroll
      for (int t = 0; t < 4; ++t) acc[t] = (v8f)0.f;
    }
    const v16bf a = load_a_frag(&Qs[cur][wave * 16 + lcol][0], half);
#pragma unroll
    for (int t = 0; t < 4; ++t) {
      const v16bf bf = load_bT_frag(&Ks[cur][t * 16 + lcol][0], half);
      acc[t] = wmma_bf16(a, bf, acc[t]);
    }
    if ((it & 15) == 15) {  // finished one 128-query strip: online update
#pragma unroll
      for (int t = 0; t < 4; ++t) {
        float tm = -1.0e30f;
#pragma unroll
        for (int r = 0; r < 8; ++r) tm = fmaxf(tm, acc[t][r] * kScale);
        float tl = 0.f;
#pragma unroll
        for (int r = 0; r < 8; ++r) tl += __expf(acc[t][r] * kScale - tm);
        const float mnew = fmaxf(m_run[t], tm);
        l_run[t] = l_run[t] * __expf(m_run[t] - mnew) + tl * __expf(tm - mnew);
        m_run[t] = mnew;
      }
    }
    __syncthreads();
  }

  // merge wave halves (rows +8): partner lane = lane ^ 16
#pragma unroll
  for (int t = 0; t < 4; ++t) {
    const float om = __shfl_xor(m_run[t], 16, 32);
    const float ol = __shfl_xor(l_run[t], 16, 32);
    const float mnew = fmaxf(m_run[t], om);
    l_run[t] = l_run[t] * __expf(m_run[t] - mnew) + ol * __expf(om - mnew);
    m_run[t] = mnew;
  }
  if (half == 0) {
#pragma unroll
    for (int t = 0; t < 4; ++t) {
      redm[wave][t * 16 + lcol] = m_run[t];
      redl[wave][t * 16 + lcol] = l_run[t];
    }
  }
  __syncthreads();
  if (tid < 64) {
    float M = -1.0e30f, L = 0.f;
#pragma unroll
    for (int w = 0; w < 8; ++w) {
      const float m2 = redm[w][tid], l2 = redl[w][tid];
      const float mnew = fmaxf(M, m2);
      L = L * __expf(M - mnew) + l2 * __expf(m2 - mnew);
      M = mnew;
    }
    colmax[(size_t)b * kS + n0 + tid] = M;
    colrcp[(size_t)b * kS + n0 + tid] = 1.0f / L;
  }
}

// ---------------------------------------------------------------------------
// Kernel 3: out[b,q,e] = x[b,q,e] + sum_k exp(s[q,k]*scale - m_k)*r_k * x[b,k,e]
// grid = (S/128 qtiles, E/64 etiles, B), block = 256.
// ---------------------------------------------------------------------------
__global__ __launch_bounds__(256) void attn_out_kernel(
    const float* __restrict__ x, const __bf16* __restrict__ qb,
    const __bf16* __restrict__ kb, const __bf16* __restrict__ xbf,
    const float* __restrict__ colmax, const float* __restrict__ colrcp,
    float* __restrict__ out) {
  __shared__ __bf16 Qs[2][128][kP];
  __shared__ __bf16 Ks[2][32][kP];
  __shared__ __bf16 Ps[128][kP];
  __shared__ __bf16 Xt[64][kP];

  const int b = blockIdx.z;
  const int m0 = blockIdx.x * 128;
  const int e0 = blockIdx.y * 64;
  const int tid = threadIdx.x;
  const int wave = tid >> 5, lane = tid & 31;
  const int half = lane >> 4, lcol = lane & 15;
  const __bf16* qB = qb + (size_t)b * kS * kD;
  const __bf16* kB = kb + (size_t)b * kS * kD;
  const __bf16* xB = xbf + (size_t)b * kS * kE;

#if defined(HAVE_TDM)
  const TdmDesc dQ = tdm_make_desc(128);
  const TdmDesc dK = tdm_make_desc(32);
  const unsigned qsOff[2] = {lds_off_of(&Qs[0][0][0]), lds_off_of(&Qs[1][0][0])};
  const unsigned ksOff[2] = {lds_off_of(&Ks[0][0][0]), lds_off_of(&Ks[1][0][0])};
#endif

  v8f oacc[4] = {};
  for (int k0 = 0; k0 < kS; k0 += 32) {
    // ---- phase 1: s tile = q[m0:m0+128] . k[k0:k0+32]^T ----
#if defined(HAVE_TDM)
    const __bf16* qT = qB + (size_t)m0 * kD;
    const __bf16* kT = kB + (size_t)k0 * kD;
    if (tid < 32) {  // prologue for this key chunk
      tdm_issue(dQ, qsOff[0], qT);
      tdm_issue(dK, ksOff[0], kT);
    }
#endif
    v8f sacc[2] = {};
    for (int dt = 0; dt < kD; dt += 32) {
      const int cur = (dt >> 5) & 1;
#if defined(HAVE_TDM)
      if (tid < 32) {
        if (dt + 32 < kD) {
          tdm_issue(dQ, qsOff[cur ^ 1], qT + dt + 32);
          tdm_issue(dK, ksOff[cur ^ 1], kT + dt + 32);
          __builtin_amdgcn_s_wait_tensorcnt(2);
        } else {
          __builtin_amdgcn_s_wait_tensorcnt(0);
        }
      }
#else
#pragma unroll
      for (int c2 = 0; c2 < 2; ++c2) {
        const int j = tid + c2 * 256;
        const int r = j >> 2, c = (j & 3) * 8;
        *(v4u*)&Qs[cur][r][c] = *(const v4u*)&qB[(size_t)(m0 + r) * kD + dt + c];
      }
      if (tid < 128) {
        const int r = tid >> 2, c = (tid & 3) * 8;
        *(v4u*)&Ks[cur][r][c] = *(const v4u*)&kB[(size_t)(k0 + r) * kD + dt + c];
      }
#endif
      __syncthreads();

      const v16bf a = load_a_frag(&Qs[cur][wave * 16 + lcol][0], half);
#pragma unroll
      for (int t = 0; t < 2; ++t) {
        const v16bf bf = load_bT_frag(&Ks[cur][t * 16 + lcol][0], half);
        sacc[t] = wmma_bf16(a, bf, sacc[t]);
      }
      __syncthreads();
    }

    // ---- phase 2: P = exp(s*scale - m_k) * r_k -> LDS ; stage x^T values ----
#pragma unroll
    for (int t = 0; t < 2; ++t) {
      const int key = k0 + t * 16 + lcol;
      const float mcol = colmax[(size_t)b * kS + key];
      const float rl = colrcp[(size_t)b * kS + key];
#pragma unroll
      for (int r = 0; r < 8; ++r) {
        const int row = wave * 16 + r + half * 8;
        const float p = __expf(sacc[t][r] * kScale - mcol) * rl;
        Ps[row][t * 16 + lcol] = (__bf16)p;
      }
    }
    {  // Xt[e][key] <- xbf[k0+r][e0+c..]; transpose, 16B global reads
      const int r = tid >> 3, c = (tid & 7) * 8;
      union { v4u v; __bf16 h[8]; } u;
      u.v = *(const v4u*)&xB[(size_t)(k0 + r) * kE + e0 + c];
#pragma unroll
      for (int i = 0; i < 8; ++i) Xt[c + i][r] = u.h[i];
    }
    if (k0 + 32 < kS)
      __builtin_prefetch(&xB[(size_t)(k0 + 32 + (tid >> 3)) * kE + e0], 0, 1);
    __syncthreads();

    // ---- phase 3: oacc += P(16x32) @ X(32x64) ----
    const v16bf pa = load_a_frag(&Ps[wave * 16 + lcol][0], half);
#pragma unroll
    for (int t = 0; t < 4; ++t) {
      const v16bf bf = load_bT_frag(&Xt[t * 16 + lcol][0], half);
      oacc[t] = wmma_bf16(pa, bf, oacc[t]);
    }
    __syncthreads();
  }

  // ---- epilogue: residual add in f32 ----
#pragma unroll
  for (int t = 0; t < 4; ++t) {
    const int col = e0 + t * 16 + lcol;
#pragma unroll
    for (int r = 0; r < 8; ++r) {
      const int row = m0 + wave * 16 + r + half * 8;
      const size_t idx = ((size_t)b * kS + row) * kE + col;
      out[idx] = x[idx] + oacc[t][r];
    }
  }
}

// ---------------------------------------------------------------------------
// Launch
// ---------------------------------------------------------------------------
extern "C" void kernel_launch(void* const* d_in, const int* in_sizes, int n_in,
                              void* d_out, int out_size, void* d_ws, size_t ws_size,
                              hipStream_t stream) {
  (void)in_sizes; (void)n_in; (void)out_size; (void)ws_size;
  const float* x  = (const float*)d_in[0];
  const float* Wq = (const float*)d_in[1];
  const float* bq = (const float*)d_in[2];
  const float* Wk = (const float*)d_in[3];
  const float* bk = (const float*)d_in[4];
  float* out = (float*)d_out;

  char* ws = (char*)d_ws;
  const size_t qkBytes = (size_t)kB * kS * kD * sizeof(__bf16);  // 16 MiB each
  __bf16* qb    = (__bf16*)(ws);
  __bf16* kb    = (__bf16*)(ws + qkBytes);
  __bf16* xbf   = (__bf16*)(ws + 2 * qkBytes);
  float* colmax = (float*)(ws + 3 * qkBytes);
  float* colrcp = (float*)(ws + 3 * qkBytes + (size_t)kB * kS * sizeof(float));

  const int nX = kB * kS * kE;
  f32_to_bf16_kernel<<<nX / (256 * 8), 256, 0, stream>>>(x, xbf);

  dim3 gProj(kS / 128, kD / 64, kB);
  proj_bias_relu_kernel<<<gProj, 256, 0, stream>>>(x, Wq, bq, qb);
  proj_bias_relu_kernel<<<gProj, 256, 0, stream>>>(x, Wk, bk, kb);

  dim3 gStats(kS / 64, kB);
  col_softmax_stats_kernel<<<gStats, 256, 0, stream>>>(qb, kb, colmax, colrcp);

  dim3 gOut(kS / 128, kE / 64, kB);
  attn_out_kernel<<<gOut, 256, 0, stream>>>(x, qb, kb, xbf, colmax, colrcp, out);
}
